// MultiHeadAttention_20040317403738
// MI455X (gfx1250) — matrix operand order repaired and checked by reading, not compiled, not run on hardware
//
#include <hip/hip_runtime.h>
#include <hip/hip_bf16.h>

// ---------------- problem constants ----------------
#define DM     768
#define HEADS  12
#define DK     64
#define SEQ    2048
#define BATCH  4
#define MROWS  (BATCH * SEQ)     // 8192 token rows
#define BH     (BATCH * HEADS)   // 48

typedef __attribute__((ext_vector_type(16))) __bf16 bf16x16;
typedef __attribute__((ext_vector_type(8)))  float  floatx8;

union Frag   { bf16x16 v; __bf16 h[16]; };
union Pack16 { __bf16 h[16]; uint4 q[2]; };

__device__ __forceinline__ floatx8 wmma_bf16(bf16x16 a, bf16x16 b, floatx8 c) {
  // v_wmma_f32_16x16x32_bf16: D = A(16x32) * B(32x16) + C(16x16 f32)
  return __builtin_amdgcn_wmma_f32_16x16x32_bf16(false, a, false, b, (short)0, c,
                                                 false, false);
}

// A fragment: 16(M) x 32(K) bf16 from LDS tile stored row-major [row][k].
// lanes 0-15: M=lane, K pairs {0,1},{2,3},{4,5},{6,7},{16,17},{18,19},{20,21},{22,23}
// lanes16-31: same M, K offset +8 (per CDNA5 ISA 16-bit A layout).
__device__ __forceinline__ bf16x16 load_a_frag(const __bf16* lds, int stride,
                                               int row0, int koff, int lane) {
  Frag f;
  const __bf16* p = lds + (row0 + (lane & 15)) * stride + koff +
                    (((lane >> 4) & 1) << 3);
  #pragma unroll
  for (int j = 0; j < 8; ++j) {
    const int k = (j < 4) ? (2 * j) : (8 + 2 * j);  // 0,2,4,6,16,18,20,22
    f.h[2 * j]     = p[k];
    f.h[2 * j + 1] = p[k + 1];
  }
  return f.v;
}

// B fragment: 32(K) x 16(N) bf16 from LDS tile stored TRANSPOSED [n][k].
// lane = N (mod 16); lanes 16-31 take K offset +8; same K pairs as the A fragment.
__device__ __forceinline__ bf16x16 load_b_frag(const __bf16* ldsT, int stride,
                                               int col0, int koff, int lane) {
  Frag f;
  const __bf16* p = ldsT + (col0 + (lane & 15)) * stride + koff +
                    (((lane >> 4) & 1) << 3);
  #pragma unroll
  for (int j = 0; j < 8; ++j) {
    const int k = (j < 4) ? (2 * j) : (8 + 2 * j);  // 0,2,4,6,16,18,20,22
    f.h[2 * j]     = p[k];
    f.h[2 * j + 1] = p[k + 1];
  }
  return f.v;
}

// =====================================================================
// Projection GEMM: C[M=8192, N=768] = A[8192,768] * W[768,768] + bias
//   A_IS_F32: A is f32 (converted to bf16 while staging) else bf16.
//   OUT_F32 : 1 -> plain f32 [M,N] output; 0 -> bf16 head-split
//             [B,H,S,dk] output, scaled by outScale (0.125 for Q).
// Workgroup tile 128x128, 8 waves (4x2), each wave 32(M)x64(N).
// Software-pipelined: tile kt+32 is fetched to registers while WMMAs
// consume tile kt from LDS (global latency overlapped with compute).
// =====================================================================
template <int A_IS_F32, int OUT_F32>
__global__ __launch_bounds__(256)
void proj_gemm(const void* __restrict__ Av, const float* __restrict__ W,
               const float* __restrict__ bias, void* __restrict__ Ov,
               float outScale) {
  constexpr int SA = 40;   // bf16 stride, 128x32 A tile (16B-aligned rows)
  constexpr int SB = 40;   // bf16 stride, transposed 128(n)x32(k) W tile
  __shared__ alignas(16) __bf16 ldsA[128 * SA];
  __shared__ alignas(16) __bf16 ldsB[128 * SB];

  const int tid  = threadIdx.x;
  const int lane = tid & 31;
  const int wv   = tid >> 5;
  const int waveRow = wv >> 1;          // 0..3  -> 32-row slice
  const int waveCol = wv & 1;           // 0..1  -> 64-col slice
  const int rowBase = blockIdx.x * 128;
  const int colBase = blockIdx.y * 128;

  const floatx8 zero = {0.f, 0.f, 0.f, 0.f, 0.f, 0.f, 0.f, 0.f};
  floatx8 acc[2][4];
  #pragma unroll
  for (int i = 0; i < 2; ++i)
    #pragma unroll
    for (int j = 0; j < 4; ++j) acc[i][j] = zero;

  const int arow  = tid >> 1;           // 0..127
  const int ahalf = tid & 1;            // 16-element half of the 32-deep row
  const int bkk   = tid >> 3;           // 0..31  (k within W tile)
  const int bnc   = (tid & 7) * 16;     // 0..112 (n chunk within W tile)

  float  aRegF[16];                     // prefetched A (f32 path)
  Pack16 aRegB;                         // prefetched A (bf16 path)
  float  bReg[16];                      // prefetched W

  auto fetch = [&](int kt) {
    if (A_IS_F32) {
      const float4* src = (const float4*)((const float*)Av +
          (size_t)(rowBase + arow) * DM + kt + ahalf * 16);
      float4 x0 = src[0], x1 = src[1], x2 = src[2], x3 = src[3];
      aRegF[0] = x0.x;  aRegF[1] = x0.y;  aRegF[2] = x0.z;  aRegF[3] = x0.w;
      aRegF[4] = x1.x;  aRegF[5] = x1.y;  aRegF[6] = x1.z;  aRegF[7] = x1.w;
      aRegF[8] = x2.x;  aRegF[9] = x2.y;  aRegF[10] = x2.z; aRegF[11] = x2.w;
      aRegF[12] = x3.x; aRegF[13] = x3.y; aRegF[14] = x3.z; aRegF[15] = x3.w;
    } else {
      const uint4* src = (const uint4*)((const __bf16*)Av +
          (size_t)(rowBase + arow) * DM + kt + ahalf * 16);
      aRegB.q[0] = src[0];
      aRegB.q[1] = src[1];
    }
    const float4* ws = (const float4*)(W + (size_t)(kt + bkk) * DM +
                                       colBase + bnc);
    float4 y0 = ws[0], y1 = ws[1], y2 = ws[2], y3 = ws[3];
    bReg[0] = y0.x;  bReg[1] = y0.y;  bReg[2] = y0.z;  bReg[3] = y0.w;
    bReg[4] = y1.x;  bReg[5] = y1.y;  bReg[6] = y1.z;  bReg[7] = y1.w;
    bReg[8] = y2.x;  bReg[9] = y2.y;  bReg[10] = y2.z; bReg[11] = y2.w;
    bReg[12] = y3.x; bReg[13] = y3.y; bReg[14] = y3.z; bReg[15] = y3.w;
  };

  auto stage = [&]() {
    Pack16 pk;
    if (A_IS_F32) {
      #pragma unroll
      for (int i = 0; i < 16; ++i) pk.h[i] = (__bf16)aRegF[i];
    } else {
      pk = aRegB;
    }
    uint4* dst = (uint4*)(ldsA + arow * SA + ahalf * 16);
    dst[0] = pk.q[0];
    dst[1] = pk.q[1];
    #pragma unroll
    for (int i = 0; i < 16; ++i)
      ldsB[(bnc + i) * SB + bkk] = (__bf16)bReg[i];
  };

  fetch(0);
  for (int kt = 0; kt < DM; kt += 32) {
    __syncthreads();                 // previous tile fully consumed
    stage();
    __syncthreads();                 // tile visible to all waves
    if (kt + 32 < DM) fetch(kt + 32);  // uniform branch; overlaps with WMMAs

    const bf16x16 a0 = load_a_frag(ldsA, SA, waveRow * 32,      0, lane);
    const bf16x16 a1 = load_a_frag(ldsA, SA, waveRow * 32 + 16, 0, lane);
    #pragma unroll
    for (int c = 0; c < 4; ++c) {
      const bf16x16 bf = load_b_frag(ldsB, SB, waveCol * 64 + c * 16, 0, lane);
      acc[0][c] = wmma_bf16(a0, bf, acc[0][c]);
      acc[1][c] = wmma_bf16(a1, bf, acc[1][c]);
    }
  }

  // ---- epilogue: D layout lane=N, VGPR r = M (lane>=16 -> M+8) ----
  const int lncol = lane & 15;
  const int mrow8 = (lane >> 4) << 3;
  #pragma unroll
  for (int rt = 0; rt < 2; ++rt)
    #pragma unroll
    for (int ct = 0; ct < 4; ++ct)
      #pragma unroll
      for (int r = 0; r < 8; ++r) {
        const int m = rowBase + waveRow * 32 + rt * 16 + r + mrow8;
        const int n = colBase + waveCol * 64 + ct * 16 + lncol;
        float v = acc[rt][ct][r] + bias[n];
        if (OUT_F32) {
          ((float*)Ov)[(size_t)m * DM + n] = v;
        } else {
          v *= outScale;
          const int b = m >> 11, s = m & (SEQ - 1);   // SEQ = 2048
          const int h = n >> 6,  d = n & (DK - 1);    // DK  = 64
          ((__bf16*)Ov)[(((size_t)(b * HEADS + h)) << 17) +
                        ((size_t)s << 6) + d] = (__bf16)v;  // SEQ*DK = 1<<17
        }
      }
}

// =====================================================================
// Scores: per (b,h) tile of  S_tile = Q * K^T  (scale folded into Q).
// Q,K bf16 [S,64]; K tiles stored row-major in LDS == transposed-B layout.
// Workgroup tile 128x128; grid (16,16,48). Whole K=64 staged once.
// =====================================================================
__global__ __launch_bounds__(256)
void attn_scores(const __bf16* __restrict__ qb, const __bf16* __restrict__ kb,
                 float* __restrict__ attn) {
  constexpr int SQ = 72;   // 64 + 8 pad, rows stay 16B aligned (144B stride)
  __shared__ alignas(16) __bf16 ldsQ[128 * SQ];
  __shared__ alignas(16) __bf16 ldsK[128 * SQ];

  const int tid  = threadIdx.x;
  const int lane = tid & 31;
  const int wv   = tid >> 5;
  const int waveRow = wv >> 1;
  const int waveCol = wv & 1;
  const int bh      = blockIdx.z;
  const int rowBase = blockIdx.x * 128;
  const int colBase = blockIdx.y * 128;
  const __bf16* Q = qb + (size_t)bh * SEQ * DK;
  const __bf16* K = kb + (size_t)bh * SEQ * DK;

  { // stage Q block [128x64] and K block [128x64] (b128 copies)
    const int row = tid >> 1, half = tid & 1;
    const uint4* qs = (const uint4*)(Q + (size_t)(rowBase + row) * DK + half * 32);
    uint4* qd = (uint4*)(ldsQ + row * SQ + half * 32);
    qd[0] = qs[0]; qd[1] = qs[1]; qd[2] = qs[2]; qd[3] = qs[3];
    const uint4* kv = (const uint4*)(K + (size_t)(colBase + row) * DK + half * 32);
    uint4* kd = (uint4*)(ldsK + row * SQ + half * 32);
    kd[0] = kv[0]; kd[1] = kv[1]; kd[2] = kv[2]; kd[3] = kv[3];
  }
  __syncthreads();

  const floatx8 zero = {0.f, 0.f, 0.f, 0.f, 0.f, 0.f, 0.f, 0.f};
  floatx8 acc[2][4];
  #pragma unroll
  for (int i = 0; i < 2; ++i)
    #pragma unroll
    for (int j = 0; j < 4; ++j) acc[i][j] = zero;

  #pragma unroll
  for (int ks = 0; ks < DK; ks += 32) {
    const bf16x16 a0 = load_a_frag(ldsQ, SQ, waveRow * 32,      ks, lane);
    const bf16x16 a1 = load_a_frag(ldsQ, SQ, waveRow * 32 + 16, ks, lane);
    #pragma unroll
    for (int c = 0; c < 4; ++c) {
      const bf16x16 bf = load_b_frag(ldsK, SQ, waveCol * 64 + c * 16, ks, lane);
      acc[0][c] = wmma_bf16(a0, bf, acc[0][c]);
      acc[1][c] = wmma_bf16(a1, bf, acc[1][c]);
    }
  }

  float* dst = attn + (size_t)bh * SEQ * SEQ;
  const int lncol = lane & 15;
  const int mrow8 = (lane >> 4) << 3;
  #pragma unroll
  for (int rt = 0; rt < 2; ++rt)
    #pragma unroll
    for (int ct = 0; ct < 4; ++ct)
      #pragma unroll
      for (int r = 0; r < 8; ++r) {
        const int m = rowBase + waveRow * 32 + rt * 16 + r + mrow8;
        const int n = colBase + waveCol * 64 + ct * 16 + lncol;
        dst[(size_t)m * SEQ + n] = acc[rt][ct][r];
      }
}

// =====================================================================
// Row softmax in place: one block (256 threads) per 2048-wide row.
// =====================================================================
__global__ __launch_bounds__(256)
void softmax_rows(float* __restrict__ attn) {
  float* p = attn + (size_t)blockIdx.x * SEQ;
  const int tid = threadIdx.x, lane = tid & 31, wv = tid >> 5;

  float4 x0 = ((const float4*)p)[tid * 2];
  float4 x1 = ((const float4*)p)[tid * 2 + 1];
  float v[8] = {x0.x, x0.y, x0.z, x0.w, x1.x, x1.y, x1.z, x1.w};

  float mx = v[0];
  #pragma unroll
  for (int i = 1; i < 8; ++i) mx = fmaxf(mx, v[i]);
  #pragma unroll
  for (int o = 16; o > 0; o >>= 1) mx = fmaxf(mx, __shfl_xor(mx, o, 32));

  __shared__ float rmax[8];
  __shared__ float rsum[8];
  if (lane == 0) rmax[wv] = mx;
  __syncthreads();
  float bm = rmax[0];
  #pragma unroll
  for (int i = 1; i < 8; ++i) bm = fmaxf(bm, rmax[i]);

  float e[8], s = 0.f;
  #pragma unroll
  for (int i = 0; i < 8; ++i) { e[i] = __expf(v[i] - bm); s += e[i]; }
  #pragma unroll
  for (int o = 16; o > 0; o >>= 1) s += __shfl_xor(s, o, 32);
  if (lane == 0) rsum[wv] = s;
  __syncthreads();
  float tot = 0.f;
  #pragma unroll
  for (int i = 0; i < 8; ++i) tot += rsum[i];
  const float inv = 1.0f / tot;

  x0 = make_float4(e[0] * inv, e[1] * inv, e[2] * inv, e[3] * inv);
  x1 = make_float4(e[4] * inv, e[5] * inv, e[6] * inv, e[7] * inv);
  ((float4*)p)[tid * 2]     = x0;
  ((float4*)p)[tid * 2 + 1] = x1;
}

// =====================================================================
// AV: per (b,h)  O = P[2048,2048](f32->bf16) * V[2048,64](bf16).
// WG tile 128(M)x64(N); 8 waves, each 16(M)x64(N).
// K staged 64-deep per iteration (32 iters, 8 WMMAs each), software-
// pipelined: next P/V tiles fetched to registers during the WMMA burst.
// Output bf16 in [B,S,H,dk] so it flattens to [8192,768] for the O-proj.
// =====================================================================
__global__ __launch_bounds__(256)
void attn_av(const float* __restrict__ attn, const __bf16* __restrict__ vb,
             __bf16* __restrict__ ab) {
  constexpr int SP = 72;   // P tile 128x64 (+8 pad, 144B rows -> 16B aligned)
  constexpr int SV = 72;   // V tile transposed 64(n)x64(k)
  __shared__ alignas(16) __bf16 ldsP[128 * SP];
  __shared__ alignas(16) __bf16 ldsV[64 * SV];

  const int tid  = threadIdx.x;
  const int lane = tid & 31;
  const int wv   = tid >> 5;         // wave -> 16-row slice
  const int bh   = blockIdx.y;
  const int b    = bh / HEADS, h = bh - b * HEADS;
  const int rowBase = blockIdx.x * 128;
  const float*  P = attn + (size_t)bh * SEQ * SEQ + (size_t)rowBase * SEQ;
  const __bf16* V = vb + (size_t)bh * SEQ * DK;

  const floatx8 zero = {0.f, 0.f, 0.f, 0.f, 0.f, 0.f, 0.f, 0.f};
  floatx8 acc[4];
  #pragma unroll
  for (int c = 0; c < 4; ++c) acc[c] = zero;

  const int prow  = tid >> 1;        // 0..127
  const int phalf = tid & 1;         // 32-element half of 64-deep row
  const int vkk   = tid >> 2;        // 0..63  (k within V tile)
  const int vnc   = (tid & 3) * 16;  // 0..48  (n chunk within V tile)

  float  pReg[32];                   // prefetched P row chunk (f32)
  Pack16 vReg;                       // prefetched V chunk (16 bf16)

  auto fetch = [&](int kt) {
    const float4* src = (const float4*)(P + (size_t)prow * SEQ + kt + phalf * 32);
    #pragma unroll
    for (int q = 0; q < 8; ++q) {
      float4 x = src[q];
      pReg[q * 4 + 0] = x.x; pReg[q * 4 + 1] = x.y;
      pReg[q * 4 + 2] = x.z; pReg[q * 4 + 3] = x.w;
    }
    const uint4* vs = (const uint4*)(V + (size_t)(kt + vkk) * DK + vnc);
    vReg.q[0] = vs[0];
    vReg.q[1] = vs[1];
  };

  auto stage = [&]() {
    Pack16 lo, hi;
    #pragma unroll
    for (int i = 0; i < 16; ++i) {
      lo.h[i] = (__bf16)pReg[i];
      hi.h[i] = (__bf16)pReg[16 + i];
    }
    uint4* dst = (uint4*)(ldsP + prow * SP + phalf * 32);
    dst[0] = lo.q[0]; dst[1] = lo.q[1];
    dst[2] = hi.q[0]; dst[3] = hi.q[1];
    #pragma unroll
    for (int i = 0; i < 16; ++i)
      ldsV[(vnc + i) * SV + vkk] = vReg.h[i];
  };

  fetch(0);
  for (int kt = 0; kt < SEQ; kt += 64) {
    __syncthreads();
    stage();
    __syncthreads();
    if (kt + 64 < SEQ) fetch(kt + 64);   // uniform; overlaps with WMMAs

    #pragma unroll
    for (int ks = 0; ks < 64; ks += 32) {
      const bf16x16 a = load_a_frag(ldsP, SP, wv * 16, ks, lane);
      #pragma unroll
      for (int c = 0; c < 4; ++c) {
        const bf16x16 bf = load_b_frag(ldsV, SV, c * 16, ks, lane);
        acc[c] = wmma_bf16(a, bf, acc[c]);
      }
    }
  }

  const int lncol = lane & 15;
  const int mrow8 = (lane >> 4) << 3;
  #pragma unroll
  for (int c = 0; c < 4; ++c)
    #pragma unroll
    for (int r = 0; r < 8; ++r) {
      const int srow = rowBase + wv * 16 + r + mrow8;
      const int n = c * 16 + lncol;
      ab[(((size_t)(b * SEQ + srow)) * HEADS + h) * DK + n] = (__bf16)acc[c][r];
    }
}

// =====================================================================
extern "C" void kernel_launch(void* const* d_in, const int* in_sizes, int n_in,
                              void* d_out, int out_size, void* d_ws,
                              size_t ws_size, hipStream_t stream) {
  (void)in_sizes; (void)n_in; (void)out_size; (void)ws_size;

  const float* query = (const float*)d_in[0];
  const float* keyx  = (const float*)d_in[1];
  const float* value = (const float*)d_in[2];
  const float* Wq = (const float*)d_in[3];
  const float* bq = (const float*)d_in[4];
  const float* Wk = (const float*)d_in[5];
  const float* bk = (const float*)d_in[6];
  const float* Wv = (const float*)d_in[7];
  const float* bv = (const float*)d_in[8];
  const float* Wo = (const float*)d_in[9];
  const float* bo = (const float*)d_in[10];

  float* out  = (float*)d_out;
  float* attn = out + (size_t)MROWS * DM;   // attn weights live in d_out

  __bf16* qb = (__bf16*)d_ws;               // [B,H,S,dk] bf16 (Q pre-scaled)
  __bf16* kb = qb + (size_t)MROWS * DM;     // [B,H,S,dk] bf16
  __bf16* vb = kb + (size_t)MROWS * DM;     // [B,H,S,dk] bf16
  __bf16* ab = vb + (size_t)MROWS * DM;     // [B,S,H,dk] bf16 attn output

  const dim3 blk(256);
  const dim3 gProj(MROWS / 128, DM / 128);        // (64, 6)

  // QKV projections (scale 1/sqrt(dk)=0.125 folded into Q)
  proj_gemm<1, 0><<<gProj, blk, 0, stream>>>(query, Wq, bq, qb, 0.125f);
  proj_gemm<1, 0><<<gProj, blk, 0, stream>>>(keyx,  Wk, bk, kb, 1.0f);
  proj_gemm<1, 0><<<gProj, blk, 0, stream>>>(value, Wv, bv, vb, 1.0f);

  // scores = Q K^T (pre-scaled), then row softmax in place
  attn_scores<<<dim3(SEQ / 128, SEQ / 128, BH), blk, 0, stream>>>(qb, kb, attn);
  softmax_rows<<<dim3(BH * SEQ), blk, 0, stream>>>(attn);

  // attn_out = P V ; out = attn_out Wo + bo
  attn_av<<<dim3(SEQ / 128, BH), blk, 0, stream>>>(attn, vb, ab);
  proj_gemm<0, 1><<<gProj, blk, 0, stream>>>(ab, Wo, bo, out, 1.0f);
}
